// WindowAttention_80968723464523
// MI455X (gfx1250) — matrix operand order repaired and checked by reading, not compiled, not run on hardware
//
#include <hip/hip_runtime.h>

typedef __attribute__((ext_vector_type(16))) _Float16 v16h;
typedef __attribute__((ext_vector_type(8)))  _Float16 v8h;
typedef __attribute__((ext_vector_type(8)))  float    v8f;

#define NPOS   49          // window tokens (7*7)
#define DIM    192
#define NH     6
#define HD     32          // head dim
#define QKVO   576         // 3*DIM
#define SCALE  0.17677669529663687f   // 32^-0.5
#define SSTR   52          // score row stride (f32)
#define PSTR   64          // prob row stride (f16), zero-padded cols 49..63
#define NWIN   2048        // 32 * 8 * 8 windows

__device__ __forceinline__ int imin(int a, int b) { return a < b ? a : b; }

// ---------------- WMMA fragment loaders (CDNA5 16x16x32 f16) ----------------
// A: 16x32 (MxK), row-major source, row stride rs (halves, multiple of 8).
// Lane l: row = l&15 (clamped to valid), K-half = l>>4. Two aligned 16B chunks.
__device__ __forceinline__ v16h load_A16(const _Float16* base, int rs,
                                         int m0, int k0, int mMax) {
  const int l   = threadIdx.x & 31;
  const int row = imin(m0 + (l & 15), mMax - 1);     // clamp: pads land in dead rows
  const _Float16* p = base + row * rs + k0 + ((l >> 4) << 4);
  const v8h lo = *(const v8h*)p;
  const v8h hi = *(const v8h*)(p + 8);
  return __builtin_shufflevector(lo, hi, 0, 1, 2, 3, 4, 5, 6, 7,
                                 8, 9, 10, 11, 12, 13, 14, 15);
}

// B: 32x16 (KxN), K-contiguous source: B[k][n] = base[n*ns + k].
__device__ __forceinline__ v16h load_B16(const _Float16* base, int ns,
                                         int k0, int n0, int nMax) {
  const int l = threadIdx.x & 31;
  const int n = imin(n0 + (l & 15), nMax - 1);       // clamp: pads land in dead cols
  const _Float16* p = base + n * ns + k0 + ((l >> 4) << 4);
  const v8h lo = *(const v8h*)p;
  const v8h hi = *(const v8h*)(p + 8);
  return __builtin_shufflevector(lo, hi, 0, 1, 2, 3, 4, 5, 6, 7,
                                 8, 9, 10, 11, 12, 13, 14, 15);
}

// B fragment from f16 QKV weight (576x192 row-major; col o = s*192 + h*32 + dd)
__device__ __forceinline__ v16h load_Bw_qkv(const _Float16* w16, int head,
                                            int s, int tn, int k0) {
  const int l = threadIdx.x & 31;
  const int o = s * DIM + head * HD + (tn & 1) * 16 + (l & 15);
  const _Float16* p = w16 + (size_t)o * DIM + k0 + ((l >> 4) << 4);
  const v8h lo = *(const v8h*)p;
  const v8h hi = *(const v8h*)(p + 8);
  return __builtin_shufflevector(lo, hi, 0, 1, 2, 3, 4, 5, 6, 7,
                                 8, 9, 10, 11, 12, 13, 14, 15);
}

__device__ __forceinline__ v8f wmma_f16(v16h a, v16h b, v8f c) {
  return __builtin_amdgcn_wmma_f32_16x16x32_f16(false, a, false, b, (short)0, c,
                                                false, false);
}

// ---------------- prep: weights -> f16, relative-position bias matrix --------
__global__ __launch_bounds__(256) void swin_prep_kernel(
    const float* __restrict__ qkv_w, const float* __restrict__ proj_w,
    const float* __restrict__ table,
    _Float16* __restrict__ qkvw16, _Float16* __restrict__ projw16,
    float* __restrict__ biasmat) {
  const int i = blockIdx.x * 256 + threadIdx.x;
  if (i < QKVO * DIM) qkvw16[i] = (_Float16)qkv_w[i];
  if (i < DIM * DIM)  projw16[i] = (_Float16)proj_w[i];
  if (i < NH * NPOS * NPOS) {
    const int h = i / (NPOS * NPOS);
    const int r = i % (NPOS * NPOS);
    const int a = r / NPOS, b = r % NPOS;
    const int idx = (a / 7 - b / 7 + 6) * 13 + (a % 7 - b % 7 + 6);
    biasmat[i] = table[idx * NH + h];
  }
}

// ---------------- fused window attention: 1 block (8 wave32) per window -----
__global__ __launch_bounds__(256) void swin_attn_kernel(
    const float* __restrict__ x, const float* __restrict__ mask,
    const float* __restrict__ qkv_b, const float* __restrict__ proj_b,
    const _Float16* __restrict__ qkvw16, const _Float16* __restrict__ projw16,
    const float* __restrict__ biasmat, float* __restrict__ out) {
  // LDS: 65,184 bytes (static). Row 49 of ctx/q/k/S is a write-dump row:
  // out-of-range tile lanes store there unconditionally; nothing reads it.
  __shared__ __align__(16) _Float16 s_xf [NPOS * DIM];        // 18,816
  __shared__ __align__(16) _Float16 s_ctx[(NPOS + 1) * DIM];  // 19,200
  __shared__ __align__(16) _Float16 s_q  [(NPOS + 1) * HD];   //  3,200
  __shared__ __align__(16) _Float16 s_k  [(NPOS + 1) * HD];   //  3,200
  __shared__ __align__(16) _Float16 s_vt [HD * PSTR];         //  4,096
  __shared__ __align__(16) float    s_S  [(NPOS + 1) * SSTR]; // 10,400
  __shared__ __align__(16) _Float16 s_P  [NPOS * PSTR];       //  6,272

  const int w   = blockIdx.x;
  const int tid = threadIdx.x;
  // wave id is uniform across the wave: pin it to an SGPR so all tile
  // bookkeeping and the q/k/v select become SALU + scalar branches.
  const int wid = __builtin_amdgcn_readfirstlane(tid >> 5);
  const int l   = tid & 31;

  // warm weight lines (global_prefetch_b8)
  if (wid == 0) {
    __builtin_prefetch(qkvw16 + (size_t)l * 3456, 0, 0);
    __builtin_prefetch(projw16 + (size_t)l * 1152, 0, 0);
  }

  // zero the K-padding that feeds valid outputs (P cols >=49; vT rows >=49)
  for (int i = tid; i < NPOS * PSTR; i += 256) s_P[i] = (_Float16)0.0f;
  for (int i = tid; i < HD * PSTR; i += 256)   s_vt[i] = (_Float16)0.0f;

  // ---- Stage A: gather window into LDS, row-major, faithful reshape order ----
  // xf[m][ch] = F[ch*49+m]; F[j] = x[b, j%192, wi*7 + (j/192)/7, wj*7 + (j/192)%7]
  {
    const int b  = w >> 6;
    const int wi = (w & 63) >> 3;
    const int wj = w & 7;
    for (int i = tid; i < NPOS * DIM; i += 256) {
      const int m = i / DIM, ch = i % DIM;
      const int j = ch * NPOS + m;
      const int c = j % DIM, p = j / DIM;
      const int hh = wi * 7 + p / 7, ww = wj * 7 + p % 7;
      s_xf[i] = (_Float16)x[(((size_t)b * DIM + c) * 56 + hh) * 56 + ww];
    }
  }
  __syncthreads();

  // ---- Per-head pipeline ----
  for (int h = 0; h < NH; ++h) {
    // B1: q|k|v = xf @ Wqkv[head cols]^T + b  (24 tiles = 3 per wave, K=192)
#pragma unroll
    for (int it = 0; it < 3; ++it) {
      const int t  = it * 8 + wid;                     // uniform trip count
      const int m0 = (t / 6) * 16, tn = t % 6;
      const int s  = tn >> 1;                          // 0=q 1=k 2=v, scalar
      v8f acc = {};
#pragma unroll
      for (int ks = 0; ks < 6; ++ks) {
        v16h a = load_A16(s_xf, DIM, m0, ks * 32, NPOS);
        v16h b = load_Bw_qkv(qkvw16, h, s, tn, ks * 32);
        acc = wmma_f16(a, b, acc);
      }
      const int dd  = (tn & 1) * 16 + (l & 15);
      const float bia = qkv_b[s * DIM + h * HD + dd];
      if (s == 0) {                                    // q, pre-scaled
#pragma unroll
        for (int r = 0; r < 8; ++r) {
          const int row = imin(m0 + r + 8 * (l >> 4), NPOS);
          s_q[row * HD + dd] = (_Float16)((acc[r] + bia) * SCALE);
        }
      } else if (s == 1) {                             // k
#pragma unroll
        for (int r = 0; r < 8; ++r) {
          const int row = imin(m0 + r + 8 * (l >> 4), NPOS);
          s_k[row * HD + dd] = (_Float16)(acc[r] + bia);
        }
      } else {                                         // v, transposed (d, m)
#pragma unroll
        for (int r = 0; r < 8; ++r) {
          const int row = m0 + r + 8 * (l >> 4);       // rows 49..63 in-bounds,
          s_vt[dd * PSTR + row] = (_Float16)(acc[r] + bia); // finite, killed by P==0
        }
      }
    }
    __syncthreads();

    // B2: S = q @ k^T  (16 tiles = 2 per wave, single K=32 WMMA each)
#pragma unroll
    for (int it = 0; it < 2; ++it) {
      const int t  = it * 8 + wid;
      const int m0 = (t >> 2) * 16, n0 = (t & 3) * 16;
      v8f acc = {};
      v16h a = load_A16(s_q, HD, m0, 0, NPOS);
      v16h b = load_B16(s_k, HD, 0, n0, NPOS);         // B[k][n] = k[n][k]
      acc = wmma_f16(a, b, acc);
      const int col = imin(n0 + (l & 15), NPOS);
#pragma unroll
      for (int r = 0; r < 8; ++r) {
        const int row = imin(m0 + r + 8 * (l >> 4), NPOS);
        s_S[row * SSTR + col] = acc[r];                // dump cell absorbs pads
      }
    }
    __syncthreads();

    // B3: softmax rows (bias + mask; q already scaled)
    if (tid < NPOS) {
      const float* bh = biasmat + ((size_t)h * NPOS + tid) * NPOS;
      const float* mr = mask + ((size_t)(w & 63) * NPOS + tid) * NPOS;
      float mx = -1e30f;
      for (int jj = 0; jj < NPOS; ++jj) {
        const float v = s_S[tid * SSTR + jj] + bh[jj] + mr[jj];
        s_S[tid * SSTR + jj] = v;
        mx = fmaxf(mx, v);
      }
      float sum = 0.f;
      for (int jj = 0; jj < NPOS; ++jj) {
        const float e = __expf(s_S[tid * SSTR + jj] - mx);
        s_S[tid * SSTR + jj] = e;
        sum += e;
      }
      const float inv = 1.0f / sum;
      for (int jj = 0; jj < NPOS; ++jj)
        s_P[tid * PSTR + jj] = (_Float16)(s_S[tid * SSTR + jj] * inv);
    }
    __syncthreads();

    // B4: ctx[:, h*32:+32] = P @ V  (8 tiles, 1 per wave, K=64 via zero tails)
    {
      const int m0 = (wid >> 1) * 16, n0 = (wid & 1) * 16;
      v8f acc = {};
#pragma unroll
      for (int ks = 0; ks < 2; ++ks) {
        v16h a = load_A16(s_P, PSTR, m0, ks * 32, NPOS);
        v16h b = load_B16(s_vt, PSTR, ks * 32, n0, HD); // B[k][n] = vT[n][k]
        acc = wmma_f16(a, b, acc);
      }
      const int col = n0 + (l & 15);
#pragma unroll
      for (int r = 0; r < 8; ++r) {
        const int row = imin(m0 + r + 8 * (l >> 4), NPOS);
        s_ctx[row * DIM + h * HD + col] = (_Float16)acc[r];
      }
    }
    __syncthreads();
  }

  // ---- Stage C: out = ctx @ Wproj^T + b  (48 tiles = 6 per wave, K=192) ----
#pragma unroll
  for (int it = 0; it < 6; ++it) {
    const int t  = it * 8 + wid;
    const int m0 = (t / 12) * 16, jn = t % 12;
    v8f acc = {};
#pragma unroll
    for (int ks = 0; ks < 6; ++ks) {
      v16h a = load_A16(s_ctx, DIM, m0, ks * 32, NPOS);
      v16h b = load_B16(projw16, DIM, ks * 32, jn * 16, DIM);  // B[k][o]=Wp[o][k]
      acc = wmma_f16(a, b, acc);
    }
    const int o = jn * 16 + (l & 15);
    const float bia = proj_b[o];
    float* op = out + (size_t)w * NPOS * DIM;
#pragma unroll
    for (int r = 0; r < 8; ++r) {
      const int row = m0 + r + 8 * (l >> 4);
      if (row < NPOS) op[row * DIM + o] = acc[r] + bia;   // HBM store keeps guard
    }
  }
}

extern "C" void kernel_launch(void* const* d_in, const int* in_sizes, int n_in,
                              void* d_out, int out_size, void* d_ws, size_t ws_size,
                              hipStream_t stream) {
  const float* x      = (const float*)d_in[0];
  const float* mask   = (const float*)d_in[1];
  const float* qkv_w  = (const float*)d_in[2];
  const float* qkv_b  = (const float*)d_in[3];
  const float* proj_w = (const float*)d_in[4];
  const float* proj_b = (const float*)d_in[5];
  const float* table  = (const float*)d_in[6];
  float* out = (float*)d_out;

  char* ws = (char*)d_ws;
  _Float16* qkvw16  = (_Float16*)(ws);                                   // 221,184 B
  _Float16* projw16 = (_Float16*)(ws + (size_t)QKVO * DIM * 2);          //  73,728 B
  float*    biasmat = (float*)(ws + (size_t)QKVO * DIM * 2 + (size_t)DIM * DIM * 2);

  swin_prep_kernel<<<(QKVO * DIM + 255) / 256, 256, 0, stream>>>(
      qkv_w, proj_w, table, qkvw16, projw16, biasmat);

  swin_attn_kernel<<<NWIN, 256, 0, stream>>>(
      x, mask, qkv_b, proj_b, qkvw16, projw16, biasmat, out);
}
